// Mamba2_22101901705606
// MI455X (gfx1250) — compile-verified, matrix operand order checked by reading, not run on hardware
//
#include <hip/hip_runtime.h>
#include <hip/hip_bf16.h>
#include <math.h>

// ---------------- problem constants ----------------
#define BS      16
#define LSEQ    1024
#define DM      384          // d_model
#define DIN     768          // d_inner
#define CD      896          // conv dim
#define NH      12           // heads
#define HD      64           // head dim
#define DST     64           // d_state
#define NCH     4            // chunks per batch
#define CHK     256          // chunk length
#define NPROJ   1676         // real in_proj rows
#define NPAD    1728         // padded to 27*64
#define MTOT    (BS*LSEQ)    // 16384 rows

// ---------------- WMMA types / helpers ----------------
typedef __attribute__((ext_vector_type(16))) __bf16 v16bf;
typedef __attribute__((ext_vector_type(8)))  float  v8f;
typedef __attribute__((ext_vector_type(4)))  unsigned int v4u;
typedef __attribute__((ext_vector_type(8)))  int v8i_;
typedef __attribute__((ext_vector_type(4)))  int v4i_;

union FragBf { v16bf v; uint4 q[2]; };

__device__ inline unsigned short f2bf(float f) {
  unsigned int u = __float_as_uint(f);
  return (unsigned short)((u + 0x7FFFu + ((u >> 16) & 1u)) >> 16);
}

// Load one 16x32 bf16 WMMA operand fragment. Memory layout: 16 "rows"
// (A: M rows / B: N columns), each row k-contiguous with leading dim `ld`
// (in elements). Matches CDNA5 16-bit A/B layout:
//   lane l: row = l&15, kb = (l&16)?8:0, halves = k in [kb..kb+7] and
//   [16+kb..16+kb+7]  -> two 16-byte loads per lane (global or LDS).
__device__ inline v16bf load_frag_k8(const unsigned short* base, int ld) {
  const int lane = threadIdx.x & 31;
  const int r  = lane & 15;
  const int kb = (lane & 16) >> 1;            // 0 or 8
  const unsigned short* p = base + (size_t)r * ld + kb;
  FragBf f;
  f.q[0] = *reinterpret_cast<const uint4*>(p);
  f.q[1] = *reinterpret_cast<const uint4*>(p + 16);
  return f.v;
}

__device__ inline v8f wmma_bf16(v16bf a, v16bf b, v8f c) {
  // D = A x B + C, f32 accumulate
  return __builtin_amdgcn_wmma_f32_16x16x32_bf16(
      false, a, false, b, (short)0, c, false, false);
}

// ---------------- Tensor Data Mover: 2D bf16 tile -> LDS ----------------
// Builds a D# (ISA 8.3/8.4): group0 = {count=1, lds_addr, global_addr, type=2},
// group1 = {data_size=2B, tensor_dim0/1, tile_dim0/1, tensor_dim0_stride}.
// Groups 2/3 zero => 2D tensor. Issued once per wave; tracked by TENSORcnt.
// This toolchain exposes the 6-arg builtin (extra int32x8 group before cpol).
__device__ inline void tdm_load_2d_bf16(unsigned lds_off, const void* gptr,
                                        unsigned dim0, unsigned dim1,
                                        unsigned tile0, unsigned tile1,
                                        unsigned stride0) {
  unsigned long long ga = (unsigned long long)gptr;
  v4u g0;
  g0.x = 1u;                                            // count=1 (valid user D#)
  g0.y = lds_off;                                       // LDS byte address
  g0.z = (unsigned)(ga & 0xFFFFFFFFu);                  // global_addr[31:0]
  g0.w = (unsigned)((ga >> 32) & 0x01FFFFFFu) | 0x80000000u; // [56:32] | type=2
  v8i_ g1;
  g1[0] = (int)0x00010000u;                             // data_size=1 (2 bytes)
  g1[1] = (int)((dim0 & 0xFFFFu) << 16);                // tensor_dim0[15:0]
  g1[2] = (int)(((dim0 >> 16) & 0xFFFFu) |              // tensor_dim0[31:16]
                ((dim1 & 0xFFFFu) << 16));              // tensor_dim1[15:0]
  g1[3] = (int)(((dim1 >> 16) & 0xFFFFu) |              // tensor_dim1[31:16]
                ((tile0 & 0xFFFFu) << 16));             // tile_dim0
  g1[4] = (int)(tile1 & 0xFFFFu);                       // tile_dim1 (tile_dim2=0)
  g1[5] = (int)stride0;                                 // tensor_dim0_stride[31:0]
  g1[6] = 0;                                            // stride hi / dim1_stride
  g1[7] = 0;
  v4i_ z4 = {0, 0, 0, 0};
  v8i_ z8 = {0, 0, 0, 0, 0, 0, 0, 0};
  __builtin_amdgcn_tensor_load_to_lds(g0, g1, z4, z4, z8, 0);
}

// ---------------- prep kernels ----------------
__global__ __launch_bounds__(256) void pack_w_in_kernel(
    const float* __restrict__ w, unsigned short* __restrict__ o) {
  int idx = blockIdx.x * 256 + threadIdx.x;          // over NPAD*DM
  if (idx >= NPAD * DM) return;
  int n = idx / DM, k = idx % DM;
  float v = (n < NPROJ) ? w[(size_t)n * DM + k] : 0.f;
  o[idx] = f2bf(v);
}

__global__ __launch_bounds__(256) void pack_w_out_kernel(
    const float* __restrict__ w, unsigned short* __restrict__ o) {
  int idx = blockIdx.x * 256 + threadIdx.x;          // over DM*DIN
  if (idx >= DM * DIN) return;
  o[idx] = f2bf(w[idx]);
}

// u (B, DM, L) -> useq bf16 (B*L, DM)
__global__ __launch_bounds__(256) void transpose_u_kernel(
    const float* __restrict__ u, unsigned short* __restrict__ o) {
  int idx = blockIdx.x * 256 + threadIdx.x;          // over MTOT*DM
  if (idx >= MTOT * DM) return;
  int row = idx / DM, c = idx % DM;
  int b = row >> 10, l = row & 1023;
  o[idx] = f2bf(u[((size_t)b * DM + c) * LSEQ + l]);
}

// ---------------- GEMM 1: in_proj  zx[M x NPAD] = useq @ W^T ----------------
// Shared 64-col weight tile staged once per block via TDM (48KB LDS),
// consumed by all 8 waves from LDS. `guard` is always 0 at runtime; the
// conditional store makes the TDM-written LDS visible to the compiler so
// the ds_load fragment reads are not folded as undef.
__global__ __launch_bounds__(256) void gemm_in_proj_kernel(
    const unsigned short* __restrict__ A, const unsigned short* __restrict__ Bw,
    float* __restrict__ C, int guard) {
  __shared__ unsigned short bsm[64 * DM];            // [n][k], ld = DM
  const int wave = threadIdx.x >> 5;
  const int m0 = blockIdx.x * 128 + wave * 16;
  const int n0 = blockIdx.y * 64;
  if (guard) bsm[threadIdx.x] = (unsigned short)guard;   // never taken
  if (wave == 0) {
    tdm_load_2d_bf16(0u, Bw + (size_t)n0 * DM,
                     /*dim0=*/DM, /*dim1=*/64,
                     /*tile0=*/DM, /*tile1=*/64, /*stride0=*/DM);
    __builtin_amdgcn_s_wait_tensorcnt(0);
  }
  __syncthreads();
  v8f acc[4] = {};
  for (int kk = 0; kk < DM; kk += 32) {
    v16bf a = load_frag_k8(A + (size_t)m0 * DM + kk, DM);
#pragma unroll
    for (int f = 0; f < 4; ++f) {
      v16bf bm = load_frag_k8(bsm + (size_t)(f * 16) * DM + kk, DM);
      acc[f] = wmma_bf16(a, bm, acc[f]);
    }
  }
  const int lane = threadIdx.x & 31;
  const int cn = lane & 15, mb = (lane & 16) >> 1;
#pragma unroll
  for (int f = 0; f < 4; ++f)
#pragma unroll
    for (int r = 0; r < 8; ++r)
      C[(size_t)(m0 + mb + r) * NPAD + n0 + f * 16 + cn] = acc[f][r];
}

// ---------------- depthwise 3x3 conv + bias + silu ----------------
__global__ __launch_bounds__(256) void conv_silu_kernel(
    const float* __restrict__ zx, const float* __restrict__ cw,
    const float* __restrict__ cb, float* __restrict__ convout,
    unsigned short* __restrict__ Bb, unsigned short* __restrict__ Cb) {
  const int row = blockIdx.x;            // b*1024 + l
  const int b = row >> 10;
  const int l = row & 1023;
  const int hh = l >> 5, ww = l & 31;
  for (int ch = threadIdx.x; ch < CD; ch += 256) {
    float s = cb[ch];
#pragma unroll
    for (int kh = 0; kh < 3; ++kh) {
      int yy = hh + kh - 1;
      if ((unsigned)yy >= 32u) continue;
#pragma unroll
      for (int kw = 0; kw < 3; ++kw) {
        int xx = ww + kw - 1;
        if ((unsigned)xx >= 32u) continue;
        s += zx[(size_t)(b * 1024 + yy * 32 + xx) * NPAD + DIN + ch] *
             cw[ch * 9 + kh * 3 + kw];
      }
    }
    float v = s / (1.f + __expf(-s));    // silu
    convout[(size_t)row * CD + ch] = v;
    if (ch >= DIN) {
      unsigned short bv = f2bf(v);
      if (ch < DIN + DST) Bb[(size_t)row * DST + (ch - DIN)] = bv;
      else                Cb[(size_t)row * DST + (ch - DIN - DST)] = bv;
    }
  }
}

// ---------------- dt softplus + per-chunk cumsum of dA ----------------
__global__ __launch_bounds__(256) void cumsum_kernel(
    const float* __restrict__ zx, const float* __restrict__ dt_bias,
    const float* __restrict__ A_log, float* __restrict__ dts,
    float* __restrict__ dAcs) {
  __shared__ float buf[CHK];
  const int bc = blockIdx.x;             // b*4 + c
  const int i = threadIdx.x;
  const int row = (bc >> 2) * 1024 + (bc & 3) * 256 + i;
  for (int h = 0; h < NH; ++h) {
    float x = zx[(size_t)row * NPAD + (DIN + CD) + h] + dt_bias[h];
    float dt = (x > 20.f) ? x : log1pf(__expf(x));
    dts[(size_t)row * NH + h] = dt;
    buf[i] = dt * (-__expf(A_log[h]));
    __syncthreads();
    for (int off = 1; off < CHK; off <<= 1) {   // inclusive Hillis-Steele
      float add = (i >= off) ? buf[i - off] : 0.f;
      __syncthreads();
      buf[i] += add;
      __syncthreads();
    }
    dAcs[(size_t)row * NH + h] = buf[i];
    __syncthreads();
  }
}

// ------- build transposed bf16 operands: xT, xwT (decay*dt folded), Cw, BmT -------
__global__ __launch_bounds__(256) void build_inputs_kernel(
    const float* __restrict__ convout, const float* __restrict__ dts,
    const float* __restrict__ dAcs,
    unsigned short* __restrict__ xT, unsigned short* __restrict__ xwT,
    unsigned short* __restrict__ Cw, unsigned short* __restrict__ BmT) {
  const int blk = blockIdx.x;            // (b*4+c)*12 + h
  const int h = blk % NH;
  const int bc = blk / NH;
  const int j = threadIdx.x;
  const int row = (bc >> 2) * 1024 + (bc & 3) * 256 + j;
  const int rowLast = (bc >> 2) * 1024 + (bc & 3) * 256 + 255;
  const float da = dAcs[(size_t)row * NH + h];
  const float wEnd = __expf(dAcs[(size_t)rowLast * NH + h] - da) *
                     dts[(size_t)row * NH + h];
  const float eA = __expf(da);
  for (int p = 0; p < HD; ++p) {
    float xv = convout[(size_t)row * CD + h * HD + p];
    xT [((size_t)blk * HD + p) * CHK + j] = f2bf(xv);
    xwT[((size_t)blk * HD + p) * CHK + j] = f2bf(xv * wEnd);
  }
  for (int n = 0; n < DST; ++n) {
    float cv = convout[(size_t)row * CD + DIN + DST + n];   // C
    Cw[((size_t)blk * CHK + j) * DST + n] = f2bf(cv * eA);
    if (h == 0) {
      float bv = convout[(size_t)row * CD + DIN + n];       // B
      BmT[((size_t)bc * DST + n) * CHK + j] = f2bf(bv);
    }
  }
}

// ---------------- GEMM 2: S[b,c][256x256] = C . B^T  (head independent) ----------------
__global__ __launch_bounds__(256) void gemm_S_kernel(
    const unsigned short* __restrict__ Cb, const unsigned short* __restrict__ Bb,
    float* __restrict__ S) {
  const int bc = blockIdx.x;
  const int wave = threadIdx.x >> 5;
  const int i0 = blockIdx.z * 128 + wave * 16;
  const int j0 = blockIdx.y * 64;
  const int base = (bc >> 2) * 1024 + (bc & 3) * 256;
  v8f acc[4] = {};
  for (int kk = 0; kk < DST; kk += 32) {
    v16bf a = load_frag_k8(Cb + (size_t)(base + i0) * DST + kk, DST);
#pragma unroll
    for (int f = 0; f < 4; ++f) {
      v16bf bm = load_frag_k8(Bb + (size_t)(base + j0 + f * 16) * DST + kk, DST);
      acc[f] = wmma_bf16(a, bm, acc[f]);
    }
  }
  const int lane = threadIdx.x & 31;
  const int cn = lane & 15, mb = (lane & 16) >> 1;
#pragma unroll
  for (int f = 0; f < 4; ++f)
#pragma unroll
    for (int r = 0; r < 8; ++r)
      S[(size_t)bc * CHK * CHK + (size_t)(i0 + mb + r) * CHK + j0 + f * 16 + cn] =
          acc[f][r];
}

// ---------------- att[b,c,h][i][j] = S * exp(dAcs_i - dAcs_j) * dt_j * causal ----------------
__global__ __launch_bounds__(256) void build_att_kernel(
    const float* __restrict__ S, const float* __restrict__ dts,
    const float* __restrict__ dAcs, unsigned short* __restrict__ att) {
  __shared__ float sda[CHK], sdt[CHK];
  const int blk = blockIdx.x; const int h = blk % NH; const int bc = blk / NH;
  const int i = threadIdx.x;
  const int row = (bc >> 2) * 1024 + (bc & 3) * 256 + i;
  sda[i] = dAcs[(size_t)row * NH + h];
  sdt[i] = dts[(size_t)row * NH + h];
  __syncthreads();
  const float dai = sda[i];
  const float* Srow = S + (size_t)bc * CHK * CHK + (size_t)i * CHK;
  unsigned short* arow = att + (size_t)blk * CHK * CHK + (size_t)i * CHK;
  for (int j = 0; j < CHK; ++j) {
    float v = 0.f;
    if (j <= i) v = Srow[j] * __expf(dai - sda[j]) * sdt[j];
    arow[j] = f2bf(v);
  }
}

// ---------------- GEMM 3: states[b,c,h][p][n] = xwT @ BmT^T  (K=256) ----------------
__global__ __launch_bounds__(128) void gemm_states_kernel(
    const unsigned short* __restrict__ xwT, const unsigned short* __restrict__ BmT,
    float* __restrict__ states) {
  const int blk = blockIdx.x;            // (b*4+c)*12 + h
  const int bc = blk / NH;
  const int wave = threadIdx.x >> 5;
  const int p0 = wave * 16;
  const unsigned short* Abase = xwT + ((size_t)blk * HD + p0) * CHK;
  const unsigned short* Bbase = BmT + (size_t)bc * DST * CHK;
  v8f acc[4] = {};
  for (int kk = 0; kk < CHK; kk += 32) {
    v16bf a = load_frag_k8(Abase + kk, CHK);
#pragma unroll
    for (int f = 0; f < 4; ++f) {
      v16bf bm = load_frag_k8(Bbase + (size_t)(f * 16) * CHK + kk, CHK);
      acc[f] = wmma_bf16(a, bm, acc[f]);
    }
  }
  const int lane = threadIdx.x & 31;
  const int cn = lane & 15, mb = (lane & 16) >> 1;
  float* Sout = states + (size_t)blk * HD * DST;
#pragma unroll
  for (int f = 0; f < 4; ++f)
#pragma unroll
    for (int r = 0; r < 8; ++r)
      Sout[(size_t)(p0 + mb + r) * DST + f * 16 + cn] = acc[f][r];
}

// ---------------- inter-chunk recurrence (4 steps, register scan) ----------------
__global__ __launch_bounds__(256) void recurrence_kernel(
    const float* __restrict__ states, const float* __restrict__ dAcs,
    unsigned short* __restrict__ prev) {
  const int b = blockIdx.x / NH, h = blockIdx.x % NH;
  const int t = threadIdx.x;
  float carry[16];
#pragma unroll
  for (int e = 0; e < 16; ++e) carry[e] = 0.f;
  for (int c = 0; c < NCH; ++c) {
    size_t base = (((size_t)(b * 4 + c) * NH) + h) * (HD * DST) + (size_t)t * 16;
#pragma unroll
    for (int e = 0; e < 16; ++e) prev[base + e] = f2bf(carry[e]);
    float dec = __expf(dAcs[(size_t)(b * 1024 + c * 256 + 255) * NH + h]);
#pragma unroll
    for (int e = 0; e < 16; ++e)
      carry[e] = carry[e] * dec + states[base + e];
  }
}

// ------- GEMM 4 (fused): y[i][p] = att@xT^T (K=256) + Cw@prev^T (K=64) + D*x -------
__global__ __launch_bounds__(256) void gemm_y_kernel(
    const unsigned short* __restrict__ att, const unsigned short* __restrict__ xT,
    const unsigned short* __restrict__ Cw, const unsigned short* __restrict__ prev,
    const float* __restrict__ convout, const float* __restrict__ Dp,
    float* __restrict__ y) {
  const int blk = blockIdx.x; const int h = blk % NH; const int bc = blk / NH;
  const int wave = threadIdx.x >> 5;
  const int i0 = blockIdx.y * 128 + wave * 16;
  v8f acc[4] = {};
  const unsigned short* Aatt = att + (size_t)blk * CHK * CHK + (size_t)i0 * CHK;
  const unsigned short* BxT  = xT + (size_t)blk * HD * CHK;
  for (int kk = 0; kk < CHK; kk += 32) {        // intra-chunk part
    v16bf a = load_frag_k8(Aatt + kk, CHK);
#pragma unroll
    for (int f = 0; f < 4; ++f) {
      v16bf bm = load_frag_k8(BxT + (size_t)(f * 16) * CHK + kk, CHK);
      acc[f] = wmma_bf16(a, bm, acc[f]);
    }
  }
  const unsigned short* Acw = Cw + (size_t)blk * CHK * DST + (size_t)i0 * DST;
  const unsigned short* Bpv = prev + (size_t)blk * HD * DST;
  for (int kk = 0; kk < DST; kk += 32) {        // inter-chunk part
    v16bf a = load_frag_k8(Acw + kk, DST);
#pragma unroll
    for (int f = 0; f < 4; ++f) {
      v16bf bm = load_frag_k8(Bpv + (size_t)(f * 16) * DST + kk, DST);
      acc[f] = wmma_bf16(a, bm, acc[f]);
    }
  }
  const int lane = threadIdx.x & 31;
  const int cn = lane & 15, mb = (lane & 16) >> 1;
  const float Dh = Dp[h];
  const int rowBase = (bc >> 2) * 1024 + (bc & 3) * 256;
#pragma unroll
  for (int f = 0; f < 4; ++f)
#pragma unroll
    for (int r = 0; r < 8; ++r) {
      int i = i0 + mb + r;
      int p = f * 16 + cn;
      int row = rowBase + i;
      float xv = convout[(size_t)row * CD + h * HD + p];
      y[(size_t)row * DIN + h * HD + p] = acc[f][r] + Dh * xv;
    }
}

// ---------------- layernorm * z  -> bf16 ----------------
__global__ __launch_bounds__(256) void layernorm_kernel(
    const float* __restrict__ y, const float* __restrict__ zx,
    const float* __restrict__ nw, const float* __restrict__ nb,
    unsigned short* __restrict__ yn) {
  __shared__ float red[256];
  const int row = blockIdx.x, t = threadIdx.x;
  const float* yr = y + (size_t)row * DIN;
  float s = 0.f;
  for (int c = t; c < DIN; c += 256) s += yr[c];
  red[t] = s; __syncthreads();
  for (int o = 128; o > 0; o >>= 1) { if (t < o) red[t] += red[t + o]; __syncthreads(); }
  float mu = red[0] / DIN; __syncthreads();
  float s2 = 0.f;
  for (int c = t; c < DIN; c += 256) { float d = yr[c] - mu; s2 += d * d; }
  red[t] = s2; __syncthreads();
  for (int o = 128; o > 0; o >>= 1) { if (t < o) red[t] += red[t + o]; __syncthreads(); }
  float rstd = rsqrtf(red[0] / DIN + 1e-5f);
  for (int c = t; c < DIN; c += 256) {
    float v = (yr[c] - mu) * rstd * nw[c] + nb[c];
    v *= zx[(size_t)row * NPAD + c];              // gate z (raw, per reference)
    yn[(size_t)row * DIN + c] = f2bf(v);
  }
}

// ---------------- GEMM 5: out = yn @ out_w^T, scatter to (B, DM, H, W) ----------------
// Weight tile staged in two 64x384 K-panels via TDM (48KB LDS each).
__global__ __launch_bounds__(256) void gemm_out_kernel(
    const unsigned short* __restrict__ A, const unsigned short* __restrict__ Bw,
    float* __restrict__ out, int guard) {
  __shared__ unsigned short bsm[64 * 384];           // [n][k-panel], ld = 384
  const int wave = threadIdx.x >> 5;
  const int m0 = blockIdx.x * 128 + wave * 16;
  const int n0 = blockIdx.y * 64;
  if (guard) bsm[threadIdx.x] = (unsigned short)guard;   // never taken
  v8f acc[4] = {};
  for (int kp = 0; kp < DIN; kp += 384) {
    __syncthreads();                                 // previous panel consumed
    if (wave == 0) {
      tdm_load_2d_bf16(0u, Bw + (size_t)n0 * DIN + kp,
                       /*dim0=*/384, /*dim1=*/64,
                       /*tile0=*/384, /*tile1=*/64, /*stride0=*/DIN);
      __builtin_amdgcn_s_wait_tensorcnt(0);
    }
    __syncthreads();
    for (int kk = 0; kk < 384; kk += 32) {
      v16bf a = load_frag_k8(A + (size_t)m0 * DIN + kp + kk, DIN);
#pragma unroll
      for (int f = 0; f < 4; ++f) {
        v16bf bm = load_frag_k8(bsm + (size_t)(f * 16) * 384 + kk, 384);
        acc[f] = wmma_bf16(a, bm, acc[f]);
      }
    }
  }
  const int lane = threadIdx.x & 31;
  const int cn = lane & 15, mb = (lane & 16) >> 1;
#pragma unroll
  for (int f = 0; f < 4; ++f)
#pragma unroll
    for (int r = 0; r < 8; ++r) {
      int m = m0 + mb + r;
      int n = n0 + f * 16 + cn;
      int b = m >> 10, l = m & 1023;
      out[((size_t)b * DM + n) * LSEQ + l] = acc[f][r];
    }
}

// ---------------- launcher ----------------
extern "C" void kernel_launch(void* const* d_in, const int* in_sizes, int n_in,
                              void* d_out, int out_size, void* d_ws, size_t ws_size,
                              hipStream_t stream) {
  const float* u        = (const float*)d_in[0];
  const float* in_w     = (const float*)d_in[1];
  const float* conv_w   = (const float*)d_in[2];
  const float* conv_b   = (const float*)d_in[3];
  const float* dt_bias  = (const float*)d_in[4];
  const float* A_log    = (const float*)d_in[5];
  const float* Dparm    = (const float*)d_in[6];
  const float* norm_w   = (const float*)d_in[7];
  const float* norm_b   = (const float*)d_in[8];
  const float* out_w    = (const float*)d_in[9];
  float* out = (float*)d_out;

  char* p = (char*)d_ws;
  auto take = [&](size_t bytes) {
    char* r = p; p += (bytes + 255) & ~(size_t)255; return r;
  };
  unsigned short* win_bf  = (unsigned short*)take((size_t)NPAD * DM * 2);
  unsigned short* wout_bf = (unsigned short*)take((size_t)DM * DIN * 2);
  unsigned short* useq    = (unsigned short*)take((size_t)MTOT * DM * 2);
  float*          zx      = (float*)take((size_t)MTOT * NPAD * 4);
  float*          convout = (float*)take((size_t)MTOT * CD * 4);
  unsigned short* Bb      = (unsigned short*)take((size_t)MTOT * DST * 2);
  unsigned short* Cb      = (unsigned short*)take((size_t)MTOT * DST * 2);
  float*          dts     = (float*)take((size_t)MTOT * NH * 4);
  float*          dAcs    = (float*)take((size_t)MTOT * NH * 4);
  unsigned short* xT      = (unsigned short*)take((size_t)BS * NCH * NH * HD * CHK * 2);
  unsigned short* xwT     = (unsigned short*)take((size_t)BS * NCH * NH * HD * CHK * 2);
  unsigned short* Cw      = (unsigned short*)take((size_t)BS * NCH * NH * CHK * DST * 2);
  unsigned short* BmT     = (unsigned short*)take((size_t)BS * NCH * DST * CHK * 2);
  float*          S       = (float*)take((size_t)BS * NCH * CHK * CHK * 4);
  unsigned short* att     = (unsigned short*)take((size_t)BS * NCH * NH * CHK * CHK * 2);
  float*          states  = (float*)take((size_t)BS * NCH * NH * HD * DST * 4);
  unsigned short* prev    = (unsigned short*)take((size_t)BS * NCH * NH * HD * DST * 2);
  float*          ybuf    = (float*)take((size_t)MTOT * DIN * 4);
  unsigned short* yn      = (unsigned short*)take((size_t)MTOT * DIN * 2);

  pack_w_in_kernel <<<(NPAD * DM + 255) / 256, 256, 0, stream>>>(in_w, win_bf);
  pack_w_out_kernel<<<(DM * DIN + 255) / 256, 256, 0, stream>>>(out_w, wout_bf);
  transpose_u_kernel<<<(MTOT * DM + 255) / 256, 256, 0, stream>>>(u, useq);

  gemm_in_proj_kernel<<<dim3(MTOT / 128, NPAD / 64), 256, 0, stream>>>(useq, win_bf,
                                                                       zx, 0);

  conv_silu_kernel<<<MTOT, 256, 0, stream>>>(zx, conv_w, conv_b, convout, Bb, Cb);

  cumsum_kernel<<<BS * NCH, 256, 0, stream>>>(zx, dt_bias, A_log, dts, dAcs);

  build_inputs_kernel<<<BS * NCH * NH, 256, 0, stream>>>(convout, dts, dAcs,
                                                         xT, xwT, Cw, BmT);

  gemm_S_kernel<<<dim3(BS * NCH, CHK / 64, 2), 256, 0, stream>>>(Cb, Bb, S);

  build_att_kernel<<<BS * NCH * NH, 256, 0, stream>>>(S, dts, dAcs, att);

  gemm_states_kernel<<<BS * NCH * NH, 128, 0, stream>>>(xwT, BmT, states);

  recurrence_kernel<<<BS * NH, 256, 0, stream>>>(states, dAcs, prev);

  gemm_y_kernel<<<dim3(BS * NCH * NH, 2), 256, 0, stream>>>(att, xT, Cw, prev,
                                                            convout, Dparm, ybuf);

  layernorm_kernel<<<MTOT, 256, 0, stream>>>(ybuf, zx, norm_w, norm_b, yn);

  gemm_out_kernel<<<dim3(MTOT / 128, DM / 64), 256, 0, stream>>>(yn, wout_bf, out, 0);
}
